// SAGE_50757923504416
// MI455X (gfx1250) — matrix operand order repaired and checked by reading, not compiled, not run on hardware
//
#include <hip/hip_runtime.h>
#include <hip/hip_bf16.h>

// GraphSAGE on MI455X (gfx1250, wave32).
// Bottleneck: edge aggregation (~2.5 GB of L2-resident gather + f32 atomics).
// Dense parts (5.7 GFLOP) run on v_wmma_f32_16x16x32_f16 with f32 accumulate.
// Atomics forced to native global_atomic_add_f32 via inline asm (no CAS loop).

#define N_NODES 100000
#define N_EDGES 1600000
#define IN_DIM  128
#define FEAT_DIM 64
#define HID_DIM  64
#define OUT_DIM  32

typedef __attribute__((ext_vector_type(16))) _Float16 v16h;
typedef __attribute__((ext_vector_type(8)))  float    v8f;

// Native CDNA5 f32 atomic add, non-returning (tracked on STOREcnt; s_endpgm's
// implicit wait-idle retires it). Device scope for cross-WGP coherence.
static __device__ __forceinline__ void atomic_add_f32(float* p, float v) {
    asm volatile("global_atomic_add_f32 %0, %1, off scope:SCOPE_DEV"
                 :: "v"(p), "v"(v) : "memory");
}

// ---------------------------------------------------------------- utilities
__global__ void zero_kernel(float* __restrict__ p, int n) {
    int i = blockIdx.x * blockDim.x + threadIdx.x;
    if (i < n) p[i] = 0.0f;
}

__global__ void deg_kernel(const int* __restrict__ dst, float* __restrict__ deg) {
    int e = blockIdx.x * blockDim.x + threadIdx.x;
    if (e < N_EDGES) atomic_add_f32(&deg[dst[e]], 1.0f);
}

__global__ void invdeg_kernel(float* __restrict__ deg) {
    int i = blockIdx.x * blockDim.x + threadIdx.x;
    if (i < N_NODES) deg[i] = 1.0f / fmaxf(deg[i], 1.0f);
}

// ---------------------------------------------------------------- edge scatter
// 16 lanes per edge, float4 each: coalesced 256B read of h[src], 4 f32 atomics to agg[dst].
__global__ void scatter_kernel(const float* __restrict__ h, const int* __restrict__ src,
                               const int* __restrict__ dst, float* __restrict__ agg) {
    int t = blockIdx.x * blockDim.x + threadIdx.x;
    int e = t >> 4;
    if (e >= N_EDGES) return;
    int c = (t & 15) << 2;
    int s = src[e], d = dst[e];
    const float4 v = *(const float4*)(h + (size_t)s * 64 + c);
    float* p = agg + (size_t)d * 64 + c;
    atomic_add_f32(p + 0, v.x);
    atomic_add_f32(p + 1, v.y);
    atomic_add_f32(p + 2, v.z);
    atomic_add_f32(p + 3, v.w);
}

// ---------------------------------------------------------------- weight pack
// Pack [W0 ; W1] (row-major [out, k]) as f16 in the exact WMMA B-operand register
// layout (ISA 7.12.2): tile t = ks*NT+nt, lane l, half i -> K = ks*32 + (l>>4)*8 + (i&7) + (i>>3)*16,
// N = nt*16 + (l&15).  One v16h (32B) load per lane per B tile in the GEMM.
__global__ void pack_b_kernel(_Float16* __restrict__ dstp,
                              const float* __restrict__ W0, int K0,
                              const float* __restrict__ W1, int K1, int Ncols) {
    int e = blockIdx.x * blockDim.x + threadIdx.x;
    int total = (K0 + K1) * Ncols;
    if (e >= total) return;
    int i    = e & 15;
    int lane = (e >> 4) & 31;
    int t    = e >> 9;
    int NT   = Ncols >> 4;
    int ks   = t / NT;
    int nt   = t - ks * NT;
    int k    = ks * 32 + ((lane >> 4) << 3) + (i & 7) + ((i >> 3) << 4);
    int o    = nt * 16 + (lane & 15);
    float v = (k < K0) ? W0[(size_t)o * K0 + k] : W1[(size_t)o * K1 + (k - K0)];
    dstp[e] = (_Float16)v;
}

// ---------------------------------------------------------------- fused SAGE GEMM
// out[n, 0:NT*16] = [scale*A0 | A1](n, 0:128) @ Bpack + bias   (+ optional ReLU)
// One wave per 16-row stripe; K = 4 steps of 32; A fragment reused across NT tiles.
template<int NT, bool RELU, bool HASA1>
__global__ __launch_bounds__(256)
void sage_gemm_kernel(const float* __restrict__ A0, int lda0,
                      const float* __restrict__ scale,
                      const float* __restrict__ A1,
                      const v16h* __restrict__ Bp,
                      const float* __restrict__ bias,
                      float* __restrict__ out) {
    const int wave = threadIdx.x >> 5;
    const int lane = threadIdx.x & 31;
    const int rowBlock = (blockIdx.x * 8 + wave) * 16;
    if (rowBlock >= N_NODES) return;           // uniform per wave
    const int m  = lane & 15;
    const int hi = lane >> 4;
    const int row = rowBlock + m;
    const bool rowOK = row < N_NODES;

    float s = 1.0f;
    if (scale) s = rowOK ? scale[row] : 0.0f;

    v8f acc[NT] = {};

#pragma unroll
    for (int ks = 0; ks < 4; ++ks) {
        // A fragment (16x32 f16), ISA layout: lane hi-half holds K = hi*8+j and 16+hi*8+j.
        v16h a = {};
        if (rowOK) {
            const float* srcp;
            float mult;
            if (HASA1 && ks >= 2) { srcp = A1 + (size_t)row * 64 + (ks - 2) * 32; mult = 1.0f; }
            else                  { srcp = A0 + (size_t)row * lda0 + ks * 32;     mult = s;   }
            const float4 f0 = *(const float4*)(srcp + hi * 8);
            const float4 f1 = *(const float4*)(srcp + hi * 8 + 4);
            const float4 f2 = *(const float4*)(srcp + hi * 8 + 16);
            const float4 f3 = *(const float4*)(srcp + hi * 8 + 20);
            a[0]  = (_Float16)(f0.x * mult); a[1]  = (_Float16)(f0.y * mult);
            a[2]  = (_Float16)(f0.z * mult); a[3]  = (_Float16)(f0.w * mult);
            a[4]  = (_Float16)(f1.x * mult); a[5]  = (_Float16)(f1.y * mult);
            a[6]  = (_Float16)(f1.z * mult); a[7]  = (_Float16)(f1.w * mult);
            a[8]  = (_Float16)(f2.x * mult); a[9]  = (_Float16)(f2.y * mult);
            a[10] = (_Float16)(f2.z * mult); a[11] = (_Float16)(f2.w * mult);
            a[12] = (_Float16)(f3.x * mult); a[13] = (_Float16)(f3.y * mult);
            a[14] = (_Float16)(f3.z * mult); a[15] = (_Float16)(f3.w * mult);
        }
#pragma unroll
        for (int nt = 0; nt < NT; ++nt) {
            v16h b = Bp[(ks * NT + nt) * 32 + lane];   // 32B/lane, L2-resident weights
            acc[nt] = __builtin_amdgcn_wmma_f32_16x16x32_f16(
                false, a, false, b, (short)0, acc[nt], false, false);
        }
    }

    // C/D layout: VGPR j, lane -> row = rowBlock + hi*8 + j, col = nt*16 + (lane&15)
#pragma unroll
    for (int nt = 0; nt < NT; ++nt) {
        const int o = nt * 16 + m;
        const float bv = bias[o];
#pragma unroll
        for (int j = 0; j < 8; ++j) {
            const int rr = rowBlock + hi * 8 + j;
            if (rr < N_NODES) {
                float v = acc[nt][j] + bv;
                if (RELU) v = fmaxf(v, 0.0f);
                out[(size_t)rr * (NT * 16) + o] = v;
            }
        }
    }
}

// ---------------------------------------------------------------- row L2 normalize
__global__ void l2norm_kernel(float* __restrict__ out) {
    const int lane = threadIdx.x & 31;
    const int row = blockIdx.x * (blockDim.x >> 5) + (threadIdx.x >> 5);
    if (row >= N_NODES) return;
    float v = out[(size_t)row * OUT_DIM + lane];
    float ss = v * v;
    ss += __shfl_xor(ss, 16, 32);
    ss += __shfl_xor(ss, 8, 32);
    ss += __shfl_xor(ss, 4, 32);
    ss += __shfl_xor(ss, 2, 32);
    ss += __shfl_xor(ss, 1, 32);
    float n = sqrtf(ss);
    out[(size_t)row * OUT_DIM + lane] = v / fmaxf(n, 1e-12f);
}

// ---------------------------------------------------------------- launcher
extern "C" void kernel_launch(void* const* d_in, const int* in_sizes, int n_in,
                              void* d_out, int out_size, void* d_ws, size_t ws_size,
                              hipStream_t stream) {
    (void)in_sizes; (void)n_in; (void)out_size; (void)ws_size;
    const float* x     = (const float*)d_in[0];
    const int*   edge  = (const int*)d_in[1];
    const int*   srcI  = edge;
    const int*   dstI  = edge + N_EDGES;
    const float* W_pre = (const float*)d_in[2];
    const float* b_pre = (const float*)d_in[3];
    const float* Wl1   = (const float*)d_in[4];
    const float* bl1   = (const float*)d_in[5];
    const float* Wr1   = (const float*)d_in[6];
    const float* Wl2   = (const float*)d_in[7];
    const float* bl2   = (const float*)d_in[8];
    const float* Wr2   = (const float*)d_in[9];
    const float* Wl3   = (const float*)d_in[10];
    const float* bl3   = (const float*)d_in[11];
    const float* Wr3   = (const float*)d_in[12];

    char* ws = (char*)d_ws;
    const size_t HB = (size_t)N_NODES * 64 * sizeof(float);      // 25.6 MB
    float*    h0   = (float*)(ws);                               // pre out / layer2 out
    float*    h1   = (float*)(ws + HB);                          // layer1 out
    float*    agg  = (float*)(ws + 2 * HB);                      // scatter target
    float*    inv  = (float*)(ws + 3 * HB);                      // deg -> inv_deg (400 KB)
    _Float16* pPre = (_Float16*)(ws + 3 * HB + (512u << 10));    // packed weights (32B aligned)
    _Float16* p1   = pPre + 128 * 64;
    _Float16* p2   = p1   + 128 * 64;
    _Float16* p3   = p2   + 128 * 64;

    const int B = 256;
    const int GN  = (N_NODES + B - 1) / B;
    const int GE  = (N_EDGES + B - 1) / B;
    const int GA  = (N_NODES * 64 + B - 1) / B;
    const int GS  = (N_EDGES * 16 + B - 1) / B;
    const int GG  = (N_NODES + 127) / 128;     // 8 waves/block * 16 rows/wave
    const int GL2 = (N_NODES + 7) / 8;

    // degrees -> inv_deg
    zero_kernel  <<<GN, B, 0, stream>>>(inv, N_NODES);
    deg_kernel   <<<GE, B, 0, stream>>>(dstI, inv);
    invdeg_kernel<<<GN, B, 0, stream>>>(inv);

    // pack weights into WMMA B layout (f16)
    pack_b_kernel<<<(128 * 64 + B - 1) / B, B, 0, stream>>>(pPre, W_pre, 128, nullptr, 0, 64);
    pack_b_kernel<<<(128 * 64 + B - 1) / B, B, 0, stream>>>(p1, Wl1, 64, Wr1, 64, 64);
    pack_b_kernel<<<(128 * 64 + B - 1) / B, B, 0, stream>>>(p2, Wl2, 64, Wr2, 64, 64);
    pack_b_kernel<<<(128 * 32 + B - 1) / B, B, 0, stream>>>(p3, Wl3, 64, Wr3, 64, 32);

    // h0 = x @ W_pre.T + b_pre
    sage_gemm_kernel<4, false, false><<<GG, B, 0, stream>>>(
        x, IN_DIM, nullptr, nullptr, (const v16h*)pPre, b_pre, h0);

    // layer 1: h1 = relu( (inv*sum h0[src]) @ Wl1.T + bl1 + h0 @ Wr1.T )
    zero_kernel   <<<GA, B, 0, stream>>>(agg, N_NODES * 64);
    scatter_kernel<<<GS, B, 0, stream>>>(h0, srcI, dstI, agg);
    sage_gemm_kernel<4, true, true><<<GG, B, 0, stream>>>(
        agg, 64, inv, h0, (const v16h*)p1, bl1, h1);

    // layer 2: h2 (reuses h0 buffer)
    zero_kernel   <<<GA, B, 0, stream>>>(agg, N_NODES * 64);
    scatter_kernel<<<GS, B, 0, stream>>>(h1, srcI, dstI, agg);
    sage_gemm_kernel<4, true, true><<<GG, B, 0, stream>>>(
        agg, 64, inv, h1, (const v16h*)p2, bl2, h0);

    // layer 3 -> d_out (unnormalized), then L2 normalize in place
    zero_kernel   <<<GA, B, 0, stream>>>(agg, N_NODES * 64);
    scatter_kernel<<<GS, B, 0, stream>>>(h0, srcI, dstI, agg);
    sage_gemm_kernel<2, false, true><<<GG, B, 0, stream>>>(
        agg, 64, inv, h0, (const v16h*)p3, bl3, (float*)d_out);
    l2norm_kernel<<<GL2, B, 0, stream>>>((float*)d_out);
}
